// EGAT_att_38130719654508
// MI455X (gfx1250) — matrix operand order repaired and checked by reading, not compiled, not run on hardware
//
#include <hip/hip_runtime.h>
#include <hip/hip_bf16.h>

// ---------------------------------------------------------------------------
// EGAT attention: out = segment_softmax( MLP([x|g]) ), MLP = 620->512->512->1
// bf16 WMMA GEMMs, TDM (tensor_load_to_lds) triple-buffered B staging,
// register-level B-fragment and A-fragment prefetch.
// ---------------------------------------------------------------------------

typedef __attribute__((ext_vector_type(16))) __bf16 v16bf;
typedef __attribute__((ext_vector_type(8)))  __bf16 v8bf;
typedef __attribute__((ext_vector_type(8)))  float  v8f;
typedef __attribute__((ext_vector_type(4)))  unsigned int v4u;
typedef __attribute__((ext_vector_type(8)))  int v8i_t;
typedef __attribute__((ext_vector_type(4)))  int v4i_t;

#define DIM  512
#define GLB  108
#define KCAT 620
#define KPAD 640
#define NSEG 1024

// LDS B tile: 256 columns x (32 data + 8 pad) bf16 -> 80B row stride
#define BROW 40
#define BTILE (256 * BROW)

static __device__ __forceinline__ void atomicMaxFloat(float* addr, float val) {
  if (val >= 0.0f) atomicMax((int*)addr, __float_as_int(val));
  else             atomicMin((unsigned int*)addr, __float_as_uint(val));
}

// --------------------------- TDM tile load (2D) ----------------------------
// Loads a [tile_d1 rows x tile_d0 elems] bf16 tile (row stride = stride0
// elements in global) into LDS at lds_off, inserting 16B of padding after
// every 64B so LDS rows land on an 80B stride (bank-conflict-free frags).
static __device__ __forceinline__ void tdm_load_tile(
    unsigned int lds_off, unsigned long long gaddr,
    unsigned int tensor_d0, unsigned int tensor_d1,
    unsigned int stride0, unsigned int tile_d0, unsigned int tile_d1)
{
  v4u g0 = {};
  g0[0] = 1u;                                          // count = 1
  g0[1] = lds_off;                                     // lds_addr (bytes)
  g0[2] = (unsigned int)(gaddr & 0xFFFFFFFFull);       // global_addr lo
  g0[3] = (unsigned int)((gaddr >> 32) & 0x01FFFFFFull) | (2u << 30); // type=2

  v8i_t g1 = {};
  g1[0] = (int)((1u << 16)        // data_size = 1 -> 2 bytes
              | (1u << 20)        // pad_enable
              | (3u << 22)        // pad_interval: 16 DWORDs (64B)
              | (3u << 25));      // pad_amount:    4 DWORDs (16B)
  g1[1] = (int)((tensor_d0 & 0xFFFFu) << 16);
  g1[2] = (int)((tensor_d0 >> 16) | ((tensor_d1 & 0xFFFFu) << 16));
  g1[3] = (int)((tensor_d1 >> 16) | (tile_d0 << 16));
  g1[4] = (int)tile_d1;                                // tile_dim2 = 0
  g1[5] = (int)stride0;                                // dim0 stride lo
  g1[6] = 0;
  g1[7] = 0;

  v4i_t zz = {};
  v8i_t z8 = {};
  __builtin_amdgcn_tensor_load_to_lds(g0, g1, zz, zz, z8, 0);
}

// ------------------------------ prep kernels -------------------------------

__global__ void k_prep_w1t(const float* __restrict__ W1, __bf16* __restrict__ W1T) {
  int i = blockIdx.x * blockDim.x + threadIdx.x;
  if (i >= DIM * KPAD) return;
  int n = i / KPAD, k = i % KPAD;
  float v = (k < KCAT) ? W1[(size_t)k * DIM + n] : 0.0f;
  W1T[i] = (__bf16)v;
}

__global__ void k_prep_w2t(const float* __restrict__ W2, __bf16* __restrict__ W2T) {
  int i = blockIdx.x * blockDim.x + threadIdx.x;
  if (i >= DIM * DIM) return;
  int n = i / DIM, k = i % DIM;
  W2T[i] = (__bf16)W2[(size_t)k * DIM + n];
}

__global__ void k_init(float* __restrict__ smax, float* __restrict__ ssum,
                       float* __restrict__ logits, int N) {
  int i = blockIdx.x * blockDim.x + threadIdx.x;
  if (i < NSEG) {
    ((unsigned int*)smax)[i] = 0xFF800000u;  // -inf
    ssum[i] = 0.0f;
  }
  if (i < N) logits[i] = 0.0f;
}

// ------------------------ A-fragment gather helpers ------------------------

static __device__ __forceinline__ v16bf load_a_cat(
    const float* __restrict__ x, const float* __restrict__ g,
    int mrow, bool valid, int ks, int klo)
{
  v16bf a;
  if (valid && (ks + 32 <= DIM)) {
    const float* rp = x + (size_t)mrow * DIM + ks;
    float4 f0 = *(const float4*)(rp + klo + 0);
    float4 f1 = *(const float4*)(rp + klo + 4);
    float4 f2 = *(const float4*)(rp + 16 + klo + 0);
    float4 f3 = *(const float4*)(rp + 16 + klo + 4);
    a[0]=(__bf16)f0.x; a[1]=(__bf16)f0.y; a[2]=(__bf16)f0.z; a[3]=(__bf16)f0.w;
    a[4]=(__bf16)f1.x; a[5]=(__bf16)f1.y; a[6]=(__bf16)f1.z; a[7]=(__bf16)f1.w;
    a[8]=(__bf16)f2.x; a[9]=(__bf16)f2.y; a[10]=(__bf16)f2.z; a[11]=(__bf16)f2.w;
    a[12]=(__bf16)f3.x; a[13]=(__bf16)f3.y; a[14]=(__bf16)f3.z; a[15]=(__bf16)f3.w;
  } else {
#pragma unroll
    for (int j = 0; j < 16; ++j) {
      int k = ks + ((j < 8) ? (klo + j) : (16 + klo + (j - 8)));
      float v = 0.0f;
      if (valid) {
        if (k < DIM) v = x[(size_t)mrow * DIM + k];
        else if (k < KCAT) v = g[(size_t)mrow * GLB + (k - DIM)];
      }
      a[j] = (__bf16)v;
    }
  }
  return a;
}

static __device__ __forceinline__ v16bf load_a_bf16(
    const __bf16* __restrict__ Y, int mrow, bool valid, int ks, int klo)
{
  v16bf a;
  if (valid) {
    const __bf16* rp = Y + (size_t)mrow * DIM + ks;
    v8bf lo  = *(const v8bf*)(rp + klo);
    v8bf hi8 = *(const v8bf*)(rp + 16 + klo);
#pragma unroll
    for (int j = 0; j < 8; ++j) { a[j] = lo[j]; a[8 + j] = hi8[j]; }
  } else {
#pragma unroll
    for (int j = 0; j < 16; ++j) a[j] = (__bf16)0.0f;
  }
  return a;
}

static __device__ __forceinline__ v16bf load_b_lds(const __bf16* bp)
{
  v8bf b0 = *(const v8bf*)(bp);
  v8bf b1 = *(const v8bf*)(bp + 16);
  v16bf b;
#pragma unroll
  for (int j = 0; j < 8; ++j) { b[j] = b0[j]; b[8 + j] = b1[j]; }
  return b;
}

// ------------------------- GEMM1: [x|g] @ W1, relu -------------------------
// Block tile: 128 rows x 256 cols. 8 waves = 4(M: 32 rows) x 2(N: 128 cols).
// Per wave: 2 M-frags x 8 N-tiles, K loop = 640/32 = 20 -> 16 WMMA per step.
// B tiles staged into LDS by TDM, triple-buffered (2 loads in flight).

__global__ __launch_bounds__(256) void k_gemm1(
    const float* __restrict__ x, const float* __restrict__ g,
    const float* __restrict__ b1, const __bf16* __restrict__ W1T,
    __bf16* __restrict__ Y1, int N)
{
  __shared__ __bf16 Bs[3][BTILE];
  constexpr int KT = 20;

  const int bm = blockIdx.x >> 1;
  const int bn = blockIdx.x & 1;
  const int lane   = threadIdx.x & 31;
  const int wave   = threadIdx.x >> 5;
  const int wm     = wave & 3;
  const int wn     = wave >> 2;
  const int lanelo = lane & 15;
  const int hi     = lane >> 4;
  const int klo    = hi * 8;
  const int kb     = hi * 8;
  const int rowbase = bm * 128 + wm * 32;
  const int mrow0 = rowbase + lanelo;
  const int mrow1 = rowbase + 16 + lanelo;
  const bool val0 = (mrow0 < N);
  const bool val1 = (mrow1 < N);

  const unsigned long long wbase =
      (unsigned long long)(size_t)W1T + (unsigned long long)(bn * 256) * KPAD * 2ull;
  const unsigned int ldsb = (unsigned int)(size_t)(&Bs[0][0]);

  if (wave == 0) {
    tdm_load_tile(ldsb + 0 * (BTILE * 2), wbase + 0ull * 64ull,
                  KPAD, DIM, KPAD, 32, 256);
    tdm_load_tile(ldsb + 1 * (BTILE * 2), wbase + 1ull * 64ull,
                  KPAD, DIM, KPAD, 32, 256);
  }

  v8f zero = {};
  v8f acc[2][8];
#pragma unroll
  for (int mf = 0; mf < 2; ++mf)
#pragma unroll
    for (int nt = 0; nt < 8; ++nt) acc[mf][nt] = zero;

  v16bf a0 = load_a_cat(x, g, mrow0, val0, 0, klo);
  v16bf a1 = load_a_cat(x, g, mrow1, val1, 0, klo);

  int ib = 0;  // LDS buffer holding tile kt
  for (int kt = 0; kt < KT; ++kt) {
    if (wave == 0) {
      if (kt + 1 < KT) __builtin_amdgcn_s_wait_tensorcnt(1);
      else             __builtin_amdgcn_s_wait_tensorcnt(0);
    }
    __syncthreads();
    if (wave == 0 && kt + 2 < KT) {
      int i2 = ib + 2; if (i2 >= 3) i2 -= 3;
      tdm_load_tile(ldsb + (unsigned int)i2 * (BTILE * 2),
                    wbase + (unsigned long long)((kt + 2) * 32) * 2ull,
                    KPAD, DIM, KPAD, 32, 256);
    }

    // prefetch next K-step's A fragments (hidden behind the WMMA burst)
    v16bf na0 = a0, na1 = a1;
    if (kt + 1 < KT) {
      na0 = load_a_cat(x, g, mrow0, val0, (kt + 1) * 32, klo);
      na1 = load_a_cat(x, g, mrow1, val1, (kt + 1) * 32, klo);
    }

    const __bf16* bs = &Bs[ib][0] + (size_t)(wn * 128 + lanelo) * BROW + kb;
    v16bf bcur = load_b_lds(bs);
#pragma unroll
    for (int nt = 0; nt < 8; ++nt) {
      v16bf bnxt = bcur;
      if (nt < 7) bnxt = load_b_lds(bs + (size_t)(nt + 1) * 16 * BROW);
      acc[0][nt] = __builtin_amdgcn_wmma_f32_16x16x32_bf16(
          false, a0, false, bcur, (short)0, acc[0][nt], false, false);
      acc[1][nt] = __builtin_amdgcn_wmma_f32_16x16x32_bf16(
          false, a1, false, bcur, (short)0, acc[1][nt], false, false);
      bcur = bnxt;
    }
    a0 = na0; a1 = na1;
    ib = (ib == 2) ? 0 : ib + 1;
  }

  // epilogue: + b1, relu, store bf16
#pragma unroll
  for (int mf = 0; mf < 2; ++mf)
#pragma unroll
    for (int nt = 0; nt < 8; ++nt) {
      const int ncol = bn * 256 + wn * 128 + nt * 16 + lanelo;
      const float bb = b1[ncol];
#pragma unroll
      for (int v = 0; v < 8; ++v) {
        int r = rowbase + mf * 16 + v + 8 * hi;
        if (r < N)
          Y1[(size_t)r * DIM + ncol] = (__bf16)fmaxf(acc[mf][nt][v] + bb, 0.0f);
      }
    }
}

// --------------- GEMM2: relu(Y1@W2+b2) fused with (.@W3) dot ---------------
// Same tiling; each block contributes a 256-column partial of the W3 dot
// product per row, accumulated with one global atomicAdd per row.

__global__ __launch_bounds__(256) void k_gemm2(
    const __bf16* __restrict__ Y1, const float* __restrict__ b2,
    const __bf16* __restrict__ W2T, const float* __restrict__ W3,
    float* __restrict__ logits, int N)
{
  __shared__ __bf16 Bs[3][BTILE];
  __shared__ float lds_logit[128];
  constexpr int KT = 16;

  if (threadIdx.x < 128) lds_logit[threadIdx.x] = 0.0f;

  const int bm = blockIdx.x >> 1;
  const int bn = blockIdx.x & 1;
  const int lane   = threadIdx.x & 31;
  const int wave   = threadIdx.x >> 5;
  const int wm     = wave & 3;
  const int wn     = wave >> 2;
  const int lanelo = lane & 15;
  const int hi     = lane >> 4;
  const int klo    = hi * 8;
  const int kb     = hi * 8;
  const int rowbase = bm * 128 + wm * 32;
  const int mrow0 = rowbase + lanelo;
  const int mrow1 = rowbase + 16 + lanelo;
  const bool val0 = (mrow0 < N);
  const bool val1 = (mrow1 < N);

  const unsigned long long wbase =
      (unsigned long long)(size_t)W2T + (unsigned long long)(bn * 256) * DIM * 2ull;
  const unsigned int ldsb = (unsigned int)(size_t)(&Bs[0][0]);

  if (wave == 0) {
    tdm_load_tile(ldsb + 0 * (BTILE * 2), wbase + 0ull * 64ull,
                  DIM, DIM, DIM, 32, 256);
    tdm_load_tile(ldsb + 1 * (BTILE * 2), wbase + 1ull * 64ull,
                  DIM, DIM, DIM, 32, 256);
  }

  v8f zero = {};
  v8f acc[2][8];
#pragma unroll
  for (int mf = 0; mf < 2; ++mf)
#pragma unroll
    for (int nt = 0; nt < 8; ++nt) acc[mf][nt] = zero;

  v16bf a0 = load_a_bf16(Y1, mrow0, val0, 0, klo);
  v16bf a1 = load_a_bf16(Y1, mrow1, val1, 0, klo);

  int ib = 0;
  for (int kt = 0; kt < KT; ++kt) {
    if (wave == 0) {
      if (kt + 1 < KT) __builtin_amdgcn_s_wait_tensorcnt(1);
      else             __builtin_amdgcn_s_wait_tensorcnt(0);
    }
    __syncthreads();
    if (wave == 0 && kt + 2 < KT) {
      int i2 = ib + 2; if (i2 >= 3) i2 -= 3;
      tdm_load_tile(ldsb + (unsigned int)i2 * (BTILE * 2),
                    wbase + (unsigned long long)((kt + 2) * 32) * 2ull,
                    DIM, DIM, DIM, 32, 256);
    }

    v16bf na0 = a0, na1 = a1;
    if (kt + 1 < KT) {
      na0 = load_a_bf16(Y1, mrow0, val0, (kt + 1) * 32, klo);
      na1 = load_a_bf16(Y1, mrow1, val1, (kt + 1) * 32, klo);
    }

    const __bf16* bs = &Bs[ib][0] + (size_t)(wn * 128 + lanelo) * BROW + kb;
    v16bf bcur = load_b_lds(bs);
#pragma unroll
    for (int nt = 0; nt < 8; ++nt) {
      v16bf bnxt = bcur;
      if (nt < 7) bnxt = load_b_lds(bs + (size_t)(nt + 1) * 16 * BROW);
      acc[0][nt] = __builtin_amdgcn_wmma_f32_16x16x32_bf16(
          false, a0, false, bcur, (short)0, acc[0][nt], false, false);
      acc[1][nt] = __builtin_amdgcn_wmma_f32_16x16x32_bf16(
          false, a1, false, bcur, (short)0, acc[1][nt], false, false);
      bcur = bnxt;
    }
    a0 = na0; a1 = na1;
    ib = (ib == 2) ? 0 : ib + 1;
  }

  // epilogue: y = relu(acc + b2[n]); partial logit = sum_n y * W3[n]
  float lacc[2][8];
#pragma unroll
  for (int mf = 0; mf < 2; ++mf)
#pragma unroll
    for (int v = 0; v < 8; ++v) lacc[mf][v] = 0.0f;

#pragma unroll
  for (int mf = 0; mf < 2; ++mf)
#pragma unroll
    for (int nt = 0; nt < 8; ++nt) {
      const int ncol = bn * 256 + wn * 128 + nt * 16 + lanelo;
      const float bb = b2[ncol];
      const float w3 = W3[ncol];
#pragma unroll
      for (int v = 0; v < 8; ++v)
        lacc[mf][v] += fmaxf(acc[mf][nt][v] + bb, 0.0f) * w3;
    }

#pragma unroll
  for (int mf = 0; mf < 2; ++mf)
#pragma unroll
    for (int v = 0; v < 8; ++v) {
      float s = lacc[mf][v];
      s += __shfl_xor(s, 1, 32);
      s += __shfl_xor(s, 2, 32);
      s += __shfl_xor(s, 4, 32);
      s += __shfl_xor(s, 8, 32);
      lacc[mf][v] = s;
    }
  if (lanelo == 0) {
#pragma unroll
    for (int mf = 0; mf < 2; ++mf)
#pragma unroll
      for (int v = 0; v < 8; ++v)
        atomicAdd(&lds_logit[wm * 32 + mf * 16 + v + 8 * hi], lacc[mf][v]);
  }
  __syncthreads();

  if (threadIdx.x < 128) {
    int row = bm * 128 + threadIdx.x;
    if (row < N) atomicAdd(&logits[row], lds_logit[threadIdx.x]);
  }
}

// ------------------------- segment softmax tail ----------------------------

__global__ void k_segmax(float* __restrict__ logits, const int* __restrict__ batch,
                         const float* __restrict__ b3, float* __restrict__ smax, int N)
{
  int i = blockIdx.x * blockDim.x + threadIdx.x;
  if (i < N) {
    float L = logits[i] + b3[0];
    logits[i] = L;
    atomicMaxFloat(&smax[batch[i]], L);
  }
}

__global__ void k_exp(const float* __restrict__ logits, const int* __restrict__ batch,
                      const float* __restrict__ smax, float* __restrict__ evals,
                      float* __restrict__ ssum, int N)
{
  int i = blockIdx.x * blockDim.x + threadIdx.x;
  if (i < N) {
    int s = batch[i];
    float e = __expf(logits[i] - smax[s]);
    evals[i] = e;
    atomicAdd(&ssum[s], e);
  }
}

__global__ void k_norm(const float* __restrict__ evals, const int* __restrict__ batch,
                       const float* __restrict__ ssum, float* __restrict__ out, int N)
{
  int i = blockIdx.x * blockDim.x + threadIdx.x;
  if (i < N) out[i] = evals[i] / (ssum[batch[i]] + 1e-16f);
}

// ------------------------------- launcher ----------------------------------

extern "C" void kernel_launch(void* const* d_in, const int* in_sizes, int n_in,
                              void* d_out, int out_size, void* d_ws, size_t ws_size,
                              hipStream_t stream)
{
  const float* x     = (const float*)d_in[0];
  const float* g     = (const float*)d_in[1];
  const int*   batch = (const int*)  d_in[2];
  const float* W1    = (const float*)d_in[3];
  const float* b1    = (const float*)d_in[4];
  const float* W2    = (const float*)d_in[5];
  const float* b2    = (const float*)d_in[6];
  const float* W3    = (const float*)d_in[7];
  const float* b3    = (const float*)d_in[8];
  float* out = (float*)d_out;

  const int N = in_sizes[0] / DIM;

  char* ws = (char*)d_ws;
  __bf16* W1T   = (__bf16*)(ws);
  __bf16* W2T   = (__bf16*)(ws + (size_t)DIM * KPAD * 2);
  __bf16* Y1    = (__bf16*)(ws + (size_t)DIM * KPAD * 2 + (size_t)DIM * DIM * 2);
  char*   after = ws + (size_t)DIM * KPAD * 2 + (size_t)DIM * DIM * 2 + (size_t)N * DIM * 2;
  float* logits = (float*)(after);
  float* evals  = (float*)(after + (size_t)N * 4);
  float* smax   = (float*)(after + (size_t)N * 8);
  float* ssum   = (float*)(after + (size_t)N * 8 + NSEG * 4);

  k_prep_w1t<<<(DIM * KPAD + 255) / 256, 256, 0, stream>>>(W1, W1T);
  k_prep_w2t<<<(DIM * DIM + 255) / 256, 256, 0, stream>>>(W2, W2T);
  k_init<<<(N + 255) / 256, 256, 0, stream>>>(smax, ssum, logits, N);

  const int nblk = ((N + 127) / 128) * 2;   // {bm} x {bn=0,1}
  k_gemm1<<<nblk, 256, 0, stream>>>(x, g, b1, W1T, Y1, N);
  k_gemm2<<<nblk, 256, 0, stream>>>(Y1, b2, W2T, W3, logits, N);

  k_segmax<<<(N + 255) / 256, 256, 0, stream>>>(logits, batch, b3, smax, N);
  k_exp   <<<(N + 255) / 256, 256, 0, stream>>>(logits, batch, smax, evals, ssum, N);
  k_norm  <<<(N + 255) / 256, 256, 0, stream>>>(evals, batch, ssum, out, N);
}
